// AttentionEncoderAdaptor_54099408060883
// MI455X (gfx1250) — matrix operand order repaired and checked by reading, not compiled, not run on hardware
//
#include <hip/hip_runtime.h>
#include <cstdint>

// Problem dims (fixed by the reference)
#define B_  2
#define S_  2048
#define D_  1024
#define H_  16
#define HD_ 64
#define BS_ (B_ * S_)

typedef __attribute__((ext_vector_type(16))) _Float16 v16h;
typedef __attribute__((ext_vector_type(8)))  _Float16 v8h;
typedef __attribute__((ext_vector_type(8)))  float    v8f;

union V16 { v16h v; v8h h[2]; };

// ---- WMMA wrapper: D = A(16x32 f16) * B(32x16 f16) + C(16x16 f32) ----
__device__ __forceinline__ v8f wmma_f16(v16h a, v16h b, v8f c) {
    return __builtin_amdgcn_wmma_f32_16x16x32_f16(
        /*neg_a=*/false, a, /*neg_b=*/false, b,
        /*c_mod=*/(short)0, c, /*reuse_a=*/false, /*reuse_b=*/false);
}

// A-matrix fragment (16x32, row-major memory). Lane l holds row (l&15).
// Per ISA layout: lanes 0-15 hold K=0..7 / 16..23, lanes 16-31 K=8..15 / 24..31.
__device__ __forceinline__ v16h load_a_frag(const _Float16* rowp, int hi) {
    V16 u;
    u.h[0] = *(const v8h*)(rowp + hi * 8);
    u.h[1] = *(const v8h*)(rowp + 16 + hi * 8);
    return u.v;
}

// B-matrix fragment (32x16). Lane l holds column (l&15); per-lane data is
// K=hi*8..hi*8+7 and 16+hi*8..16+hi*8+7 (same K order as the A fragment).
__device__ __forceinline__ v16h load_b_frag(const _Float16* colp, int hi) {
    V16 u;
    u.h[0] = *(const v8h*)(colp + hi * 8);
    u.h[1] = *(const v8h*)(colp + 16 + hi * 8);
    return u.v;
}

// ---------------- fp32 -> f16 converts ----------------
__global__ void cvt_f32_to_f16(const float* __restrict__ x,
                               _Float16* __restrict__ y, int n) {
    int i = blockIdx.x * blockDim.x + threadIdx.x;
    if (i < n) y[i] = (_Float16)x[i];
}

__global__ void cvt_weights(const float* __restrict__ wq, const float* __restrict__ wk,
                            const float* __restrict__ wv, const float* __restrict__ wo,
                            _Float16* __restrict__ y) {
    int i = blockIdx.x * blockDim.x + threadIdx.x;  // n = 4*D*D exactly
    const int DD = D_ * D_;
    int sel = i / DD, j = i - sel * DD;
    const float* src = (sel == 0) ? wq : (sel == 1) ? wk : (sel == 2) ? wv : wo;
    y[i] = (_Float16)src[j];
}

// ---------------- fused QKV projection ----------------
// y = x @ W^T + b  ;  W row n is the B-matrix column n -> contiguous loads.
// grid: (3*D/64, BS/32), block 32 (1 wave -> 32x64 output tile, 8 accums).
// Inner loop is software-pipelined: fragments for k+32 are loaded before the
// 8 WMMAs of step k, so WMMAs only wait on the previous iteration's loads.
__global__ void qkv_gemm(const _Float16* __restrict__ X,
                         const _Float16* __restrict__ W,   // [wq|wk|wv] rows, [3D, D]
                         const float* __restrict__ bq, const float* __restrict__ bk,
                         const float* __restrict__ bv,
                         _Float16* __restrict__ Q,    // [B,S,D] (pre-scaled)
                         _Float16* __restrict__ Ko,   // [B,S,D]
                         _Float16* __restrict__ Vt) { // [B,H,HD,S]
    __shared__ _Float16 tile[16 * 16];
    const int l = threadIdx.x, lane = l & 15, hi = l >> 4;
    const int m0 = blockIdx.y * 32;
    const int n0 = blockIdx.x * 64;

    v8f acc[2][4] = {};
    const _Float16* arow0 = X + (size_t)(m0 + lane) * D_;
    const _Float16* arow1 = X + (size_t)(m0 + 16 + lane) * D_;
    const _Float16* wrow  = W + (size_t)(n0 + lane) * D_;

    // pipeline prologue: k0 = 0 fragments
    v16h a0 = load_a_frag(arow0, hi);
    v16h a1 = load_a_frag(arow1, hi);
    v16h b[4];
    #pragma unroll
    for (int t = 0; t < 4; ++t)
        b[t] = load_b_frag(wrow + (size_t)t * 16 * D_, hi);

    #pragma unroll 4
    for (int k0 = 0; k0 < D_ - 32; k0 += 32) {
        // issue next-step loads first (fresh SSA values => overlap with WMMAs)
        v16h na0 = load_a_frag(arow0 + k0 + 32, hi);
        v16h na1 = load_a_frag(arow1 + k0 + 32, hi);
        v16h nb[4];
        #pragma unroll
        for (int t = 0; t < 4; ++t)
            nb[t] = load_b_frag(wrow + (size_t)t * 16 * D_ + k0 + 32, hi);
        __builtin_prefetch(arow0 + k0 + 288, 0, 3);   // -> global_prefetch_b8
        __builtin_prefetch(arow1 + k0 + 288, 0, 3);
        #pragma unroll
        for (int t = 0; t < 4; ++t) {
            acc[0][t] = wmma_f16(a0, b[t], acc[0][t]);   // B fragment reused x2
            acc[1][t] = wmma_f16(a1, b[t], acc[1][t]);
        }
        a0 = na0; a1 = na1;
        #pragma unroll
        for (int t = 0; t < 4; ++t) b[t] = nb[t];
    }
    // epilogue k-step
    #pragma unroll
    for (int t = 0; t < 4; ++t) {
        acc[0][t] = wmma_f16(a0, b[t], acc[0][t]);
        acc[1][t] = wmma_f16(a1, b[t], acc[1][t]);
    }

    const int seg   = n0 / D_;        // 0=q 1=k 2=v (64 | D, so uniform per block)
    const int ncolb = n0 - seg * D_;

    if (seg == 0) {                    // Q, scaled by HD^-0.5 = 0.125
        #pragma unroll
        for (int mh = 0; mh < 2; ++mh)
            #pragma unroll
            for (int t = 0; t < 4; ++t) {
                int ncol = ncolb + t * 16 + lane;
                float bias = bq[ncol];
                #pragma unroll
                for (int r = 0; r < 8; ++r) {
                    int srow = m0 + mh * 16 + r + 8 * hi;
                    Q[(size_t)srow * D_ + ncol] =
                        (_Float16)((acc[mh][t][r] + bias) * 0.125f);
                }
            }
    } else if (seg == 1) {             // K
        #pragma unroll
        for (int mh = 0; mh < 2; ++mh)
            #pragma unroll
            for (int t = 0; t < 4; ++t) {
                int ncol = ncolb + t * 16 + lane;
                float bias = bk[ncol];
                #pragma unroll
                for (int r = 0; r < 8; ++r) {
                    int srow = m0 + mh * 16 + r + 8 * hi;
                    Ko[(size_t)srow * D_ + ncol] = (_Float16)(acc[mh][t][r] + bias);
                }
            }
    } else {                           // V: transpose via LDS into [B,H,HD,S]
        const int bb = m0 / S_;
        #pragma unroll
        for (int mh = 0; mh < 2; ++mh) {
            const int s0loc = (m0 + mh * 16) - bb * S_;
            #pragma unroll
            for (int t = 0; t < 4; ++t) {
                float bias = bv[ncolb + t * 16 + lane];
                #pragma unroll
                for (int r = 0; r < 8; ++r)
                    tile[(r + 8 * hi) * 16 + lane] = (_Float16)(acc[mh][t][r] + bias);
                asm volatile("" ::: "memory");   // LDS ops are in-order per wave
                if (hi == 0) {
                    int dglob = ncolb + t * 16 + lane;
                    int hh = dglob >> 6, dl = dglob & 63;
                    _Float16 col[16];
                    #pragma unroll
                    for (int rr = 0; rr < 16; ++rr) col[rr] = tile[rr * 16 + lane];
                    size_t base = ((size_t)(bb * H_ + hh) * HD_ + dl) * S_ + s0loc;
                    v8h* cp = (v8h*)col;
                    *(v8h*)(Vt + base)     = cp[0];
                    *(v8h*)(Vt + base + 8) = cp[1];
                }
                asm volatile("" ::: "memory");
            }
        }
    }
}

// ---------------- flash attention ----------------
// grid: (S/16, B*H), block 32. One wave handles 16 query rows of one head.
__global__ void attn_kernel(const _Float16* __restrict__ Q,
                            const _Float16* __restrict__ K,
                            const _Float16* __restrict__ Vt,
                            const float* __restrict__ mask,   // [B,1,S,S]
                            _Float16* __restrict__ C) {       // [B,S,D]
    __shared__ _Float16 pt[16 * 32];
    const int l = threadIdx.x, lane = l & 15, hi = l >> 4;
    const int bh = blockIdx.y, b = bh / H_, h = bh - b * H_;
    const int s0 = blockIdx.x * 16;

    const _Float16* qrow = Q + (size_t)(b * S_ + s0 + lane) * D_ + h * HD_;
    v16h a0 = load_a_frag(qrow, hi);        // q cols 0..31
    v16h a1 = load_a_frag(qrow + 32, hi);   // q cols 32..63

    float mrow[8], lsum[8];
    v8f o[4] = {};
    #pragma unroll
    for (int r = 0; r < 8; ++r) { mrow[r] = -1e30f; lsum[r] = 0.0f; }

    for (int kt = 0; kt < S_; kt += 32) {
        // scores: Q(16x64) @ K_tile^T(64x32) -> two 16x16 C tiles.
        // Load all 4 K fragments up front, then issue the 4 WMMAs.
        v16h kb[2][2];
        #pragma unroll
        for (int nc = 0; nc < 2; ++nc) {
            const _Float16* krow =
                K + (size_t)(b * S_ + kt + nc * 16 + lane) * D_ + h * HD_;
            __builtin_prefetch(krow + 32 * D_, 0, 3);   // next key tile
            kb[nc][0] = load_b_frag(krow, hi);
            kb[nc][1] = load_b_frag(krow + 32, hi);
        }
        v8f sc[2] = {};
        #pragma unroll
        for (int nc = 0; nc < 2; ++nc) {
            sc[nc] = wmma_f16(a0, kb[nc][0], sc[nc]);
            sc[nc] = wmma_f16(a1, kb[nc][1], sc[nc]);
        }
        // online softmax (row = (r, hi); row values live across 16 lanes)
        float p0[8], p1[8];
        #pragma unroll
        for (int r = 0; r < 8; ++r) {
            int qrow_i = s0 + r + 8 * hi;
            const float* mp = mask + ((size_t)b * S_ + qrow_i) * S_ + kt;
            float v0 = sc[0][r] + mp[lane];
            float v1 = sc[1][r] + mp[16 + lane];
            float mx = fmaxf(v0, v1);
            #pragma unroll
            for (int off = 1; off < 16; off <<= 1)
                mx = fmaxf(mx, __shfl_xor(mx, off));
            float mnew = fmaxf(mrow[r], mx);
            float f = __expf(mrow[r] - mnew);
            mrow[r] = mnew;
            p0[r] = __expf(v0 - mnew);
            p1[r] = __expf(v1 - mnew);
            float ssum = p0[r] + p1[r];
            #pragma unroll
            for (int off = 1; off < 16; off <<= 1)
                ssum += __shfl_xor(ssum, off);
            lsum[r] = lsum[r] * f + ssum;
            #pragma unroll
            for (int nc = 0; nc < 4; ++nc) o[nc][r] *= f;
        }
        // P tile: C-layout -> A-layout via LDS (per-wave, in-order DS ops)
        #pragma unroll
        for (int r = 0; r < 8; ++r) {
            pt[(r + 8 * hi) * 32 + lane]      = (_Float16)p0[r];
            pt[(r + 8 * hi) * 32 + 16 + lane] = (_Float16)p1[r];
        }
        asm volatile("" ::: "memory");
        V16 u;
        u.h[0] = *(const v8h*)(pt + lane * 32 + hi * 8);
        u.h[1] = *(const v8h*)(pt + lane * 32 + 16 + hi * 8);
        v16h pa = u.v;
        asm volatile("" ::: "memory");
        // ctx += P(16x32) @ V_tile(32x64); Vt rows are contiguous along S
        v16h vb[4];
        #pragma unroll
        for (int nc = 0; nc < 4; ++nc) {
            const _Float16* vrow =
                Vt + ((size_t)(b * H_ + h) * HD_ + nc * 16 + lane) * S_ + kt;
            vb[nc] = load_b_frag(vrow, hi);
        }
        #pragma unroll
        for (int nc = 0; nc < 4; ++nc)
            o[nc] = wmma_f16(pa, vb[nc], o[nc]);
    }
    // normalize and store ctx back into [B,S,D] (head-sliced)
    #pragma unroll
    for (int r = 0; r < 8; ++r) {
        float inv = 1.0f / lsum[r];
        size_t base = (size_t)(b * S_ + s0 + r + 8 * hi) * D_ + h * HD_;
        #pragma unroll
        for (int nc = 0; nc < 4; ++nc)
            C[base + nc * 16 + lane] = (_Float16)(o[nc][r] * inv);
    }
}

// ---------------- output projection + gate ----------------
// gated = ctx @ Wo^T + bo ; out = features * gated
// grid: (D/64, BS/32), block 32 (1 wave -> 32x64 tile), software-pipelined.
__global__ void out_gemm(const _Float16* __restrict__ Cx,
                         const _Float16* __restrict__ Wo,
                         const float* __restrict__ bo,
                         const float* __restrict__ feat,
                         float* __restrict__ outp, float* __restrict__ gated) {
    const int l = threadIdx.x, lane = l & 15, hi = l >> 4;
    const int m0 = blockIdx.y * 32;
    const int n0 = blockIdx.x * 64;

    v8f acc[2][4] = {};
    const _Float16* arow0 = Cx + (size_t)(m0 + lane) * D_;
    const _Float16* arow1 = Cx + (size_t)(m0 + 16 + lane) * D_;
    const _Float16* wrow  = Wo + (size_t)(n0 + lane) * D_;

    v16h a0 = load_a_frag(arow0, hi);
    v16h a1 = load_a_frag(arow1, hi);
    v16h b[4];
    #pragma unroll
    for (int t = 0; t < 4; ++t)
        b[t] = load_b_frag(wrow + (size_t)t * 16 * D_, hi);

    #pragma unroll 4
    for (int k0 = 0; k0 < D_ - 32; k0 += 32) {
        v16h na0 = load_a_frag(arow0 + k0 + 32, hi);
        v16h na1 = load_a_frag(arow1 + k0 + 32, hi);
        v16h nb[4];
        #pragma unroll
        for (int t = 0; t < 4; ++t)
            nb[t] = load_b_frag(wrow + (size_t)t * 16 * D_ + k0 + 32, hi);
        __builtin_prefetch(arow0 + k0 + 288, 0, 3);
        __builtin_prefetch(arow1 + k0 + 288, 0, 3);
        #pragma unroll
        for (int t = 0; t < 4; ++t) {
            acc[0][t] = wmma_f16(a0, b[t], acc[0][t]);
            acc[1][t] = wmma_f16(a1, b[t], acc[1][t]);
        }
        a0 = na0; a1 = na1;
        #pragma unroll
        for (int t = 0; t < 4; ++t) b[t] = nb[t];
    }
    #pragma unroll
    for (int t = 0; t < 4; ++t) {
        acc[0][t] = wmma_f16(a0, b[t], acc[0][t]);
        acc[1][t] = wmma_f16(a1, b[t], acc[1][t]);
    }

    #pragma unroll
    for (int mh = 0; mh < 2; ++mh)
        #pragma unroll
        for (int t = 0; t < 4; ++t) {
            int ncol = n0 + t * 16 + lane;
            float bias = bo[ncol];
            #pragma unroll
            for (int r = 0; r < 8; ++r) {
                size_t idx = (size_t)(m0 + mh * 16 + r + 8 * hi) * D_ + ncol;
                float g = acc[mh][t][r] + bias;
                gated[idx] = g;
                outp[idx]  = feat[idx] * g;
            }
        }
}

extern "C" void kernel_launch(void* const* d_in, const int* in_sizes, int n_in,
                              void* d_out, int out_size, void* d_ws, size_t ws_size,
                              hipStream_t stream) {
    const float* feat = (const float*)d_in[0];
    const float* mask = (const float*)d_in[1];
    const float* wq = (const float*)d_in[2];
    const float* bq = (const float*)d_in[3];
    const float* wk = (const float*)d_in[4];
    const float* bk = (const float*)d_in[5];
    const float* wv = (const float*)d_in[6];
    const float* bv = (const float*)d_in[7];
    const float* wo = (const float*)d_in[8];
    const float* bo = (const float*)d_in[9];

    float* out0  = (float*)d_out;                      // out  [B,S,D]
    float* gated = out0 + (size_t)BS_ * D_;            // gated[B,S,D]

    // workspace carve-up (f16): X | Wqkv | Wo | Q | K | Vt | ctx  (~48 MB)
    _Float16* ws   = (_Float16*)d_ws;
    _Float16* X16  = ws;                 size_t off = (size_t)BS_ * D_;
    _Float16* W16  = ws + off;           off += (size_t)3 * D_ * D_;
    _Float16* Wo16 = ws + off;           off += (size_t)D_ * D_;   // contiguous after W16
    _Float16* Q16  = ws + off;           off += (size_t)BS_ * D_;
    _Float16* K16  = ws + off;           off += (size_t)BS_ * D_;
    _Float16* Vt16 = ws + off;           off += (size_t)BS_ * D_;
    _Float16* C16  = ws + off;

    const int NX = BS_ * D_;
    cvt_f32_to_f16<<<NX / 256, 256, 0, stream>>>(feat, X16, NX);
    cvt_weights<<<(4 * D_ * D_) / 256, 256, 0, stream>>>(wq, wk, wv, wo, W16);

    qkv_gemm<<<dim3(3 * D_ / 64, BS_ / 32), 32, 0, stream>>>(
        X16, W16, bq, bk, bv, Q16, K16, Vt16);

    attn_kernel<<<dim3(S_ / 16, B_ * H_), 32, 0, stream>>>(
        Q16, K16, Vt16, mask, C16);

    out_gemm<<<dim3(D_ / 64, BS_ / 32), 32, 0, stream>>>(
        C16, Wo16, bo, feat, out0, gated);
}
